// SspTransformerEmbedding_51943334477844
// MI455X (gfx1250) — matrix operand order repaired and checked by reading, not compiled, not run on hardware
//
#include <hip/hip_runtime.h>
#include <hip/hip_bf16.h>
#include <math.h>

typedef __attribute__((ext_vector_type(16))) _Float16 v16h;
typedef __attribute__((ext_vector_type(8)))  float    v8f;

#define DDIM 512
#define NPH  257
#define KP   544           // padded K: 2*257=514 -> 17*32
#define KSTEPS 17
#define AST  552           // A_lds row stride (halves), 16B-aligned, bank-friendly
#define BST  552           // B_lds row stride (halves)
#define YSTH 520           // y_lds row stride (halves, f16)
#define TWG  128           // tokens per workgroup
#define NTHREADS 256
#define PANEL_Q4 1088      // float4 per 16x544h panel
#define PANEL_Q2 2176      // float2 per 16x544h panel
#define STAGE_Q2 (2*PANEL_Q2)   // 4352 float2 per step (two panels) = 17 * 256
#define GREGS (STAGE_Q2/NTHREADS)  // 17 float2 per thread, exactly uniform

// LDS carve offsets (bytes), all 16B aligned
#define OFF_TH 0
#define OFF_X  4112                        // 1028 floats theta
#define OFF_A  6160                        // 512 floats x
#define OFF_B  (OFF_A + TWG*AST*2)         // 147472
#define BBUF_H (2*16*BST)                  // halves per B buffer (2 panels)
#define OFF_Y  (OFF_B + BBUF_H*2)          // 182800
#define SMEM_TOTAL (OFF_Y + TWG*YSTH*2)    // 315920

// ---------------------------------------------------------------------------
// Kernel 1: fold irfft into W.
//   Gt[n][2k]   =  w_k/512 * sum_d cos(2*pi*k*d/512) * W[n][d]
//   Gt[n][2k+1] = -w_k/512 * sum_d sin(2*pi*k*d/512) * W[n][d]
// stored f16, n-major with row stride KP (zero-padded to KP).
// ---------------------------------------------------------------------------
__global__ void build_gt(const float* __restrict__ W, _Float16* __restrict__ Gt) {
  __shared__ float wrow[DDIM];
  __shared__ float ctab[DDIM];
  const int tid = threadIdx.x;
  const int n = blockIdx.x;
  wrow[tid]       = W[n * DDIM + tid];
  wrow[tid + 256] = W[n * DDIM + tid + 256];
  const float step = 6.28318530717958647692f / 512.0f;
  ctab[tid]       = cosf(step * (float)tid);
  ctab[tid + 256] = cosf(step * (float)(tid + 256));
  __syncthreads();
  for (int r = tid; r < KP; r += NTHREADS) {
    float res = 0.0f;
    if (r < 2 * NPH) {
      const int k = r >> 1;
      int t = (r & 1) ? 384 : 0;      // odd rows use shifted table == sin
      float acc = 0.0f;
      for (int d = 0; d < DDIM; ++d) {
        acc = __fmaf_rn(ctab[t], wrow[d], acc);
        t = (t + k) & 511;
      }
      const float w = (k == 0 || k == 256) ? 1.0f : 2.0f;
      const float sgn = (r & 1) ? -1.0f : 1.0f;
      res = acc * (sgn * w * (1.0f / 512.0f));
    }
    Gt[(size_t)n * KP + r] = (_Float16)res;
  }
}

// ---------------------------------------------------------------------------
// Kernel 2: phasors -> WMMA GEMM [128 x 544]x[544 x 512] -> fused LayerNorm.
// Each wave owns 32 tokens (two A-fragment sets, register-resident via the
// gfx1250 extended-VGPR file): every B fragment feeds 2 WMMAs. B-panel
// staging is register-held across the WMMA chain with a perfectly uniform
// 17 x float2 per-thread split (no exec-masked tail -> no spill/early wait).
// ---------------------------------------------------------------------------
__global__ __launch_bounds__(NTHREADS, 1)
void ssp_main(const float* __restrict__ x,
              const float* __restrict__ theta,
              const _Float16* __restrict__ Gt,
              const float* __restrict__ gamma,
              float* __restrict__ out) {
  extern __shared__ char smem[];
  float*    th_lds = (float*)(smem + OFF_TH);
  float*    x_lds  = (float*)(smem + OFF_X);
  _Float16* A_lds  = (_Float16*)(smem + OFF_A);
  _Float16* B_lds  = (_Float16*)(smem + OFF_B);
  _Float16* y_lds  = (_Float16*)(smem + OFF_Y);

  const int tid  = threadIdx.x;
  const int wave = tid >> 5;
  const int lane = tid & 31;
  const int half = lane >> 4;
  const int l16  = lane & 15;
  const long tb  = (long)blockIdx.x * TWG;   // first token of this block

  // ---- stage theta (4x257) and x (128x4, contiguous) ----
  for (int i = tid; i < 4 * NPH; i += NTHREADS) th_lds[i] = theta[i];
  for (int i = tid; i < TWG * 4; i += NTHREADS) x_lds[i] = x[tb * 4 + i];
  __syncthreads();

  // ---- phasors: A_lds[t][2k]=cos(2pi*phi_k), [2k+1]=sin, f16 ----
  for (int i = tid; i < TWG * NPH; i += NTHREADS) {
    const int t = i / NPH;
    const int k = i - t * NPH;
    const float* xr = &x_lds[t * 4];
    const float phi = 2.0f * (xr[0] * th_lds[k] + xr[1] * th_lds[NPH + k] +
                              xr[2] * th_lds[2 * NPH + k] + xr[3] * th_lds[3 * NPH + k]);
    float s, c;
    __sincosf(6.28318530717958647692f * phi, &s, &c);
    A_lds[t * AST + 2 * k]     = (_Float16)c;
    A_lds[t * AST + 2 * k + 1] = (_Float16)s;
  }
  // zero K-padding columns 514..543
  for (int i = tid; i < TWG * (KP - 2 * NPH); i += NTHREADS) {
    const int t = i / (KP - 2 * NPH);
    const int c = i - t * (KP - 2 * NPH);
    A_lds[t * AST + 2 * NPH + c] = (_Float16)0.0f;
  }
  __syncthreads();

  const int mq    = wave & 3;                // 32-token chunk
  const int nhalf = wave >> 2;               // which 256-col half
  const int m0 = mq * 32;

  // ---- preload BOTH A-fragment sets (2 x 17 x v16h, extended VGPR file) ----
  union AB { v16h v; float4 q[2]; };
  AB Afa[KSTEPS], Afb[KSTEPS];
  {
    const _Float16* ra = A_lds + (m0 + l16) * AST;
    const _Float16* rb = A_lds + (m0 + 16 + l16) * AST;
#pragma unroll
    for (int kk = 0; kk < KSTEPS; ++kk) {
      Afa[kk].q[0] = *(const float4*)(ra + kk * 32 + half * 8);
      Afa[kk].q[1] = *(const float4*)(ra + kk * 32 + 16 + half * 8);
      Afb[kk].q[0] = *(const float4*)(rb + kk * 32 + half * 8);
      Afb[kk].q[1] = *(const float4*)(rb + kk * 32 + 16 + half * 8);
    }
  }

  // ---- initial B panels (nt=0 and nt=16) directly to LDS ----
  {
    const float4* s0 = (const float4*)(Gt);
    const float4* s1 = (const float4*)(Gt + (size_t)16 * 16 * KP);
    float4* d0 = (float4*)(B_lds);
    float4* d1 = d0 + 16 * (BST / 8);
    for (int i = tid; i < PANEL_Q4; i += NTHREADS) {
      const int nl = i / 68;
      const int j  = i - nl * 68;
      d0[nl * (BST / 8) + j] = s0[i];
      d1[nl * (BST / 8) + j] = s1[i];
    }
  }
  __syncthreads();

  for (int p = 0; p < 16; ++p) {
    // ---- phase G: issue global loads for step p+1 into registers (no wait,
    //      perfectly uniform 17 x float2 per thread) ----
    float2 g[GREGS];
    if (p < 15) {
      const float2* s0 = (const float2*)(Gt + (size_t)(p + 1) * 16 * KP);
      const float2* s1 = (const float2*)(Gt + (size_t)(17 + p) * 16 * KP);
#pragma unroll
      for (int u = 0; u < GREGS; ++u) {
        const int i = tid + u * NTHREADS;
        g[u] = (i < PANEL_Q2) ? s0[i] : s1[i - PANEL_Q2];
      }
      if (p < 14)  // warm following panels in L2 (global_prefetch_b8)
        __builtin_prefetch((const char*)(Gt + (size_t)(p + 2) * 16 * KP) + tid * 128, 0, 1);
    }

    // ---- compute step p: 2x17 WMMAs, each B fragment feeds two chains ----
    const _Float16* bcol = B_lds + (nhalf * 16 + l16) * BST;
    AB Bm[2];
    Bm[0].q[0] = *(const float4*)(bcol + half * 8);
    Bm[0].q[1] = *(const float4*)(bcol + 16 + half * 8);
    v8f acca = {}, accb = {};
#pragma unroll
    for (int kk = 0; kk < KSTEPS; ++kk) {
      if (kk + 1 < KSTEPS) {
        Bm[(kk + 1) & 1].q[0] = *(const float4*)(bcol + (kk + 1) * 32 + half * 8);
        Bm[(kk + 1) & 1].q[1] = *(const float4*)(bcol + (kk + 1) * 32 + 16 + half * 8);
      }
      acca = __builtin_amdgcn_wmma_f32_16x16x32_f16(false, Afa[kk].v, false, Bm[kk & 1].v,
                                                    (short)0, acca, false, false);
      accb = __builtin_amdgcn_wmma_f32_16x16x32_f16(false, Afb[kk].v, false, Bm[kk & 1].v,
                                                    (short)0, accb, false, false);
    }

    // ---- C/D layout: VGPR r -> row 8*half + r, col = nt*16 + l16 ----
    const int nt = nhalf * 16 + p;
#pragma unroll
    for (int r = 0; r < 8; ++r) {
      y_lds[(m0 + half * 8 + r) * YSTH + nt * 16 + l16]      = (_Float16)acca[r];
      y_lds[(m0 + 16 + half * 8 + r) * YSTH + nt * 16 + l16] = (_Float16)accb[r];
    }
    __syncthreads();   // all waves done reading B panel

    // ---- phase S: write staged registers to the B buffer (uniform) ----
    if (p < 15) {
#pragma unroll
      for (int u = 0; u < GREGS; ++u) {
        const int i = tid + u * NTHREADS;
        const int pa = (i >= PANEL_Q2) ? 1 : 0;
        const int j  = i - pa * PANEL_Q2;
        const int nl = j / 136;            // 136 float2 per source row
        const int jj = j - nl * 136;
        ((float2*)(B_lds + pa * 16 * BST))[nl * (BST / 4) + jj] = g[u];
      }
      __syncthreads(); // new panel visible to all waves
    }
  }

  // ---- fused LayerNorm (scale-only) + store; each wave: 16 tokens ----
  for (int tt = 0; tt < 16; ++tt) {
    const int t = wave * 16 + tt;
    float sum = 0.0f, sq = 0.0f;
    float v[16];
#pragma unroll
    for (int j = 0; j < 16; ++j) {
      const float yv = (float)y_lds[t * YSTH + lane + 32 * j];
      v[j] = yv;
      sum += yv;
      sq  += yv * yv;
    }
#pragma unroll
    for (int m = 16; m >= 1; m >>= 1) {
      sum += __shfl_xor(sum, m, 32);
      sq  += __shfl_xor(sq, m, 32);
    }
    const float mean = sum * (1.0f / 512.0f);
    const float var  = sq * (1.0f / 512.0f) - mean * mean;
    const float rstd = rsqrtf(var + 1e-5f);
    const long row = (tb + t) * DDIM;
#pragma unroll
    for (int j = 0; j < 16; ++j) {
      const int col = lane + 32 * j;
      out[row + col] = (v[j] - mean) * rstd * gamma[col];
    }
  }
}

// ---------------------------------------------------------------------------
extern "C" void kernel_launch(void* const* d_in, const int* in_sizes, int n_in,
                              void* d_out, int out_size, void* d_ws, size_t ws_size,
                              hipStream_t stream) {
  const float* x     = (const float*)d_in[0];   // [B,S,4]   f32
  const float* theta = (const float*)d_in[1];   // [4,257]   f32
  const float* W     = (const float*)d_in[2];   // [512,512] f32
  const float* gamma = (const float*)d_in[3];   // [512]     f32
  float* out = (float*)d_out;                   // [B,S,512] f32
  _Float16* Gt = (_Float16*)d_ws;               // [512][544] f16 = 544 KiB

  build_gt<<<DDIM, NTHREADS, 0, stream>>>(W, Gt);

  const int tokens  = in_sizes[0] / 4;          // B*S = 262144
  const int nblocks = tokens / TWG;             // 2048
  ssp_main<<<nblocks, NTHREADS, (size_t)SMEM_TOTAL, stream>>>(x, theta, Gt, gamma, out);
}
